// MSDeformableAttention_89885075571163
// MI455X (gfx1250) — matrix operand order repaired and checked by reading, not compiled, not run on hardware
//
#include <hip/hip_runtime.h>

// MS-deformable attention for MI455X (gfx1250), wave32 + bf16 WMMA,
// LDS-staged weight strips via gfx1250 async-to-LDS, 4-wide wave tiles.
// E=256, NH=8, NL=3, NP=4, HD=32, B=4, Lq=Lv=13125.

#define E_DIM  256
#define NHEAD  8
#define NLVL   3
#define NPTS   4
#define HDIM   32
#define LQ     13125
#define BATCH  4
#define MROWS  (BATCH * LQ)   /* 52500 */
#define KDIM   256

typedef __attribute__((ext_vector_type(16))) __bf16 v16bf;
typedef __attribute__((ext_vector_type(8)))  float  v8f;
typedef int v4i_gcc __attribute__((vector_size(16)));   // matches builtin's V4i

union FragBF {
    v16bf v;
    unsigned short u[16];
    uint4 q[2];
};

__device__ __forceinline__ unsigned short f32_to_bf16(float f) {
    unsigned int x = __builtin_bit_cast(unsigned int, f);
    x += 0x7FFFu + ((x >> 16) & 1u);   // round-to-nearest-even
    return (unsigned short)(x >> 16);
}

__device__ __forceinline__ void async_lds_copy16(const unsigned short* gsrc,
                                                 unsigned short* ldst) {
    // copy 8 bf16 (16B) global -> LDS
#if __has_builtin(__builtin_amdgcn_global_load_async_to_lds_b128)
    __builtin_amdgcn_global_load_async_to_lds_b128(
        (__attribute__((address_space(1))) v4i_gcc*)(size_t)gsrc,
        (__attribute__((address_space(3))) v4i_gcc*)(unsigned int)(size_t)ldst, 0, 0);
#else
    *(uint4*)ldst = *(const uint4*)gsrc;
#endif
}

__device__ __forceinline__ void async_lds_wait() {
#if __has_builtin(__builtin_amdgcn_global_load_async_to_lds_b128)
    asm volatile("s_wait_asynccnt 0" ::: "memory");
#endif
}

// ---------------------------------------------------------------------------
// Bulk f32 -> bf16 conversion, 8 elements per thread (b128 in, b128 out).
// ---------------------------------------------------------------------------
__global__ void msda_f32_to_bf16(const float* __restrict__ src,
                                 unsigned short* __restrict__ dst, int n8) {
    int i = blockIdx.x * blockDim.x + threadIdx.x;
    if (i >= n8) return;
    const float4 a = ((const float4*)src)[2 * i];
    const float4 b = ((const float4*)src)[2 * i + 1];
    union { unsigned short u[8]; uint4 q; } o;
    o.u[0] = f32_to_bf16(a.x); o.u[1] = f32_to_bf16(a.y);
    o.u[2] = f32_to_bf16(a.z); o.u[3] = f32_to_bf16(a.w);
    o.u[4] = f32_to_bf16(b.x); o.u[5] = f32_to_bf16(b.y);
    o.u[6] = f32_to_bf16(b.z); o.u[7] = f32_to_bf16(b.w);
    ((uint4*)dst)[i] = o.q;
}

// ---------------------------------------------------------------------------
// Weight transpose+convert: Bt[n*K + k] = bf16(W[k*N + n]), K = 256.
// ---------------------------------------------------------------------------
__global__ void msda_transpose_w(const float* __restrict__ W,
                                 unsigned short* __restrict__ Bt, int N) {
    int idx = blockIdx.x * blockDim.x + threadIdx.x;
    if (idx >= N * KDIM) return;
    int n = idx / KDIM;
    int k = idx - n * KDIM;
    Bt[(size_t)n * KDIM + k] = f32_to_bf16(W[(size_t)k * N + n]);
}

// ---------------------------------------------------------------------------
// GEMM: C(MxN,f32) = A(MxK,bf16) * B(KxN) + bias, Bt = bf16 NxK row-major.
// Block = 256 threads = 8 waves; blockIdx.y selects a strip of NT 16-col
// tiles. The 16*NT x 256 bf16 weight strip (8*NT KB) is staged in LDS once
// per block via global_load_async_to_lds_b128. Each wave computes a
// 16 x (16*NT) tile: per k-step one A-fragment load feeds NT WMMAs
// (A reuse x NT -> 4x less activation traffic for NT=4).
// A-fragment: lane m = lane&15, kh = lane>>4
//   elem e<8 : K = ks*32 + kh*8 + e ;  elem e>=8 : K = ks*32 + 16 + kh*8 + (e-8)
// B-fragment: lane n = lane&15, same K mapping as A-fragment
// ---------------------------------------------------------------------------
template <int NT>
__global__ __launch_bounds__(256) void msda_gemm_bf16(
        const unsigned short* __restrict__ A, const unsigned short* __restrict__ Bt,
        const float* __restrict__ bias, float* __restrict__ C, int M, int N) {
    __shared__ unsigned short sB[NT * 16 * KDIM];   // NT * 8 KB weight strip

    // ---- stage B strip: contiguous, NT*2 x 16B per thread -------------
    {
        const unsigned short* gbase = Bt + (size_t)blockIdx.y * (NT * 16 * KDIM);
#pragma unroll
        for (int j = 0; j < NT * 2; ++j) {
            const int e = (j * 256 + threadIdx.x) * 8;   // bf16 element index
            async_lds_copy16(gbase + e, &sB[e]);
        }
        async_lds_wait();
    }
    __syncthreads();

    const int lane   = threadIdx.x & 31;
    const int wv     = threadIdx.x >> 5;
    const int mtile  = blockIdx.x * 8 + wv;
    const int mtiles = (M + 15) >> 4;
    if (mtile < mtiles) {                       // wave-uniform; EXEC stays all-1s
        const int kh   = lane >> 4;
        const int lidx = lane & 15;

        int rowA = mtile * 16 + lidx;
        if (rowA > M - 1) rowA = M - 1;         // clamp tail loads
        const unsigned short* arow = A + (size_t)rowA * KDIM;

        v8f acc[NT];
#pragma unroll
        for (int t = 0; t < NT; ++t)
            acc[t] = (v8f){0.f, 0.f, 0.f, 0.f, 0.f, 0.f, 0.f, 0.f};

#pragma unroll
        for (int ks = 0; ks < KDIM / 32; ++ks) {
            const int ka = ks * 32 + kh * 8;    // A: two runs of 8 contiguous K
            FragBF fa;
            fa.q[0] = *(const uint4*)(arow + ka);
            fa.q[1] = *(const uint4*)(arow + ka + 16);

            const int kb = ks * 32 + kh * 8;    // B: two runs of 8 contiguous K from LDS
#pragma unroll
            for (int t = 0; t < NT; ++t) {
                FragBF fb;
                fb.q[0] = *(const uint4*)(&sB[(t * 16 + lidx) * KDIM + kb]);
                fb.q[1] = *(const uint4*)(&sB[(t * 16 + lidx) * KDIM + kb + 16]);
                acc[t] = __builtin_amdgcn_wmma_f32_16x16x32_bf16(
                             false, fa.v, false, fb.v, (short)0, acc[t], false, false);
            }
        }

#pragma unroll
        for (int t = 0; t < NT; ++t) {
            const int   col = blockIdx.y * (NT * 16) + t * 16 + lidx;
            const float bb  = bias[col];
#pragma unroll
            for (int r = 0; r < 8; ++r) {
                const int row = mtile * 16 + kh * 8 + r;   // D: M = r + 8*kh
                if (row < M) C[(size_t)row * N + col] = acc[t][r] + bb;
            }
        }
    }
}

// ---------------------------------------------------------------------------
// Softmax over NL*NP = 12 logits per (b,q,h).
// ---------------------------------------------------------------------------
__global__ void msda_softmax12(float* __restrict__ aw) {
    int t = blockIdx.x * blockDim.x + threadIdx.x;
    if (t >= MROWS * NHEAD) return;
    float* p = aw + (size_t)(t >> 3) * (NHEAD * NLVL * NPTS)
                  + (size_t)(t & 7) * (NLVL * NPTS);
    float m = p[0];
#pragma unroll
    for (int j = 1; j < 12; ++j) m = fmaxf(m, p[j]);
    float e[12];
    float s = 0.f;
#pragma unroll
    for (int j = 0; j < 12; ++j) { e[j] = __expf(p[j] - m); s += e[j]; }
    const float inv = 1.f / s;
#pragma unroll
    for (int j = 0; j < 12; ++j) p[j] = e[j] * inv;
}

// ---------------------------------------------------------------------------
// Bilinear sampling + attention-weighted sum. One wave32 per (b,q,h);
// lane = head-dim channel -> each gather is a coalesced 128B row from the
// L2-resident projected value tensor. Branch conditions are wave-uniform.
// Output written directly as bf16 (A operand of the final GEMM).
// ---------------------------------------------------------------------------
__global__ __launch_bounds__(256) void msda_sample(
        const float* __restrict__ v,      // (B*Lv, 256) row-major f32
        const float* __restrict__ refp,   // (B*Lq, NL, 2)
        const float* __restrict__ off,    // (B*Lq, NH*NL*NP*2)
        const float* __restrict__ aw,     // (B*Lq, NH*NL*NP)
        unsigned short* __restrict__ mid) // (B*Lq, 256) bf16
{
    const int lane = threadIdx.x & 31;
    const int wid  = blockIdx.x * 8 + (threadIdx.x >> 5);
    const int h    = wid & (NHEAD - 1);
    const int bq   = wid >> 3;
    if (bq >= MROWS) return;
    const int b = bq / LQ;

    const float* offr = off  + (size_t)bq * (NHEAD * NLVL * NPTS * 2)
                             + (size_t)h * (NLVL * NPTS * 2);
    const float* awr  = aw   + (size_t)bq * (NHEAD * NLVL * NPTS)
                             + (size_t)h * (NLVL * NPTS);
    const float* refr = refp + (size_t)bq * (NLVL * 2);

    const int Hls[NLVL]  = {100, 50, 25};
    const int Wls[NLVL]  = {100, 50, 25};
    const int Loff[NLVL] = {0, 10000, 12500};

    float acc = 0.f;
#pragma unroll
    for (int l = 0; l < NLVL; ++l) {
        const float rx = refr[l * 2 + 0];
        const float ry = refr[l * 2 + 1];
        const int Wl = Wls[l], Hl = Hls[l];
        const float* vb = v + ((size_t)(b * LQ + Loff[l])) * E_DIM
                            + (size_t)h * HDIM + lane;
#pragma unroll
        for (int p = 0; p < NPTS; ++p) {
            // loc*size - 0.5  ==  ref*size + off - 0.5  (norm = (Wl, Hl))
            const float x = rx * (float)Wl + offr[l * (NPTS * 2) + p * 2 + 0] - 0.5f;
            const float y = ry * (float)Hl + offr[l * (NPTS * 2) + p * 2 + 1] - 0.5f;
            const float w = awr[l * NPTS + p];
            const float xf = floorf(x), yf = floorf(y);
            const int x0 = (int)xf, y0 = (int)yf;
            const float wx1 = x - xf, wy1 = y - yf;
            const float wx0 = 1.f - wx1, wy0 = 1.f - wy1;
            const bool vx0 = (x0 >= 0) & (x0 < Wl);
            const bool vx1 = (x0 + 1 >= 0) & (x0 + 1 < Wl);
            const bool vy0 = (y0 >= 0) & (y0 < Hl);
            const bool vy1 = (y0 + 1 >= 0) & (y0 + 1 < Hl);
            if (vy0 & vx0) acc += (w * wx0 * wy0) * vb[(size_t)(y0 * Wl + x0) * E_DIM];
            if (vy0 & vx1) acc += (w * wx1 * wy0) * vb[(size_t)(y0 * Wl + x0 + 1) * E_DIM];
            if (vy1 & vx0) acc += (w * wx0 * wy1) * vb[(size_t)((y0 + 1) * Wl + x0) * E_DIM];
            if (vy1 & vx1) acc += (w * wx1 * wy1) * vb[(size_t)((y0 + 1) * Wl + x0 + 1) * E_DIM];
        }
    }
    mid[(size_t)bq * E_DIM + (size_t)h * HDIM + lane] = f32_to_bf16(acc);
}

// ---------------------------------------------------------------------------
extern "C" void kernel_launch(void* const* d_in, const int* in_sizes, int n_in,
                              void* d_out, int out_size, void* d_ws, size_t ws_size,
                              hipStream_t stream) {
    (void)in_sizes; (void)n_in; (void)out_size; (void)ws_size;
    const float* query = (const float*)d_in[0];
    const float* refp  = (const float*)d_in[1];
    const float* value = (const float*)d_in[2];
    /* d_in[3] = value_spatial_shapes: static, hardcoded */
    const float* Wv    = (const float*)d_in[4];
    const float* bv    = (const float*)d_in[5];
    const float* Woff  = (const float*)d_in[6];
    const float* boff  = (const float*)d_in[7];
    const float* Watt  = (const float*)d_in[8];
    const float* batt  = (const float*)d_in[9];
    const float* Wo    = (const float*)d_in[10];
    const float* bo    = (const float*)d_in[11];
    float* out = (float*)d_out;

    // Workspace partition (~195 MB), 256B-aligned slabs.
    char* ws = (char*)d_ws;
    size_t o = 0;
    auto take = [&](size_t bytes) -> char* {
        char* p = ws + o;
        o += (bytes + 255) & ~(size_t)255;
        return p;
    };
    float* v_proj = (float*)take((size_t)MROWS * 256 * 4);
    float* off_b  = (float*)take((size_t)MROWS * 192 * 4);
    float* aw_b   = (float*)take((size_t)MROWS * 96 * 4);
    unsigned short* q_bf   = (unsigned short*)take((size_t)MROWS * 256 * 2);
    unsigned short* val_bf = (unsigned short*)take((size_t)MROWS * 256 * 2);
    unsigned short* mid_bf = (unsigned short*)take((size_t)MROWS * 256 * 2);
    unsigned short* WvT   = (unsigned short*)take((size_t)256 * 256 * 2);
    unsigned short* WoT   = (unsigned short*)take((size_t)256 * 256 * 2);
    unsigned short* WoffT = (unsigned short*)take((size_t)192 * 256 * 2);
    unsigned short* WattT = (unsigned short*)take((size_t)96  * 256 * 2);

    // 1) activation f32->bf16 conversions + weight transpose/convert
    const int n8 = MROWS * 256 / 8;   // 1,680,000
    msda_f32_to_bf16<<<(n8 + 255) / 256, 256, 0, stream>>>(query, q_bf,   n8);
    msda_f32_to_bf16<<<(n8 + 255) / 256, 256, 0, stream>>>(value, val_bf, n8);
    msda_transpose_w<<<(256 * 256 + 255) / 256, 256, 0, stream>>>(Wv,   WvT,   256);
    msda_transpose_w<<<(256 * 256 + 255) / 256, 256, 0, stream>>>(Wo,   WoT,   256);
    msda_transpose_w<<<(192 * 256 + 255) / 256, 256, 0, stream>>>(Woff, WoffT, 192);
    msda_transpose_w<<<(96  * 256 + 255) / 256, 256, 0, stream>>>(Watt, WattT, 96);

    const int mtiles = (MROWS + 15) / 16;    // 3282
    const int gx     = (mtiles + 7) / 8;     // 8 waves per 256-thread block

    // 2) projections (bf16 WMMA, f32 accumulate); strips of 64 (or 32) cols
    msda_gemm_bf16<4><<<dim3(gx, 256 / 64), 256, 0, stream>>>(val_bf, WvT,   bv,   v_proj, MROWS, 256);
    msda_gemm_bf16<4><<<dim3(gx, 192 / 64), 256, 0, stream>>>(q_bf,   WoffT, boff, off_b,  MROWS, 192);
    msda_gemm_bf16<2><<<dim3(gx, 96  / 32), 256, 0, stream>>>(q_bf,   WattT, batt, aw_b,   MROWS, 96);

    // 3) softmax over 12 (level,point) logits per (b,q,h)
    msda_softmax12<<<(MROWS * NHEAD + 255) / 256, 256, 0, stream>>>(aw_b);

    // 4) bilinear sampling + attention-weighted reduction (L2-resident gathers)
    msda_sample<<<MROWS, 256, 0, stream>>>(v_proj, refp, off_b, aw_b, mid_bf);

    // 5) output projection straight into d_out
    msda_gemm_bf16<4><<<dim3(gx, 256 / 64), 256, 0, stream>>>(mid_bf, WoT, bo, out, MROWS, 256);
}
